// HardAttention_62689342652578
// MI455X (gfx1250) — matrix operand order repaired and checked by reading, not compiled, not run on hardware
//
#include <hip/hip_runtime.h>
#include <math.h>

// ---------------------------------------------------------------------------
// HardAttention (L>1 branch), MI455X / gfx1250
//   out  = tanh(concat(context, output) @ W^T + b)   [B,L,D]  (M=16384,N=512,K=1024)
//   attn = constant one-hot [B,L,S]
// Path A (ws large enough): pre-convert to bf16 in d_ws, then TDM
//   (tensor_load_to_lds, double-buffered, s_wait_tensorcnt) feeds
//   v_wmma_f32_16x16x32_bf16. Path B: round-1 convert-in-kernel GEMM.
// ---------------------------------------------------------------------------

typedef __attribute__((ext_vector_type(16))) __bf16        bf16x16;
typedef __attribute__((ext_vector_type(8)))  float         f32x8;
typedef __attribute__((ext_vector_type(4)))  float         f32x4;
typedef __attribute__((ext_vector_type(4)))  unsigned int  u32x4;
typedef __attribute__((ext_vector_type(8)))  int           i32x8;
typedef __attribute__((ext_vector_type(4)))  int           i32x4;

#define B_   4
#define L_   4096
#define S_   4096
#define D_   512
#define K_   1024            // 2*D
#define M_   (B_ * L_)       // 16384

#define BM   128             // block M tile
#define BN   64              // block N tile
#define LDA  40              // LDS row stride (ushorts): 32 data + 8 pad = 80 B
#define LDB  40

union FragAB {
  unsigned int   u32[8];
  unsigned short u16[16];
  bf16x16        v;
};

static __device__ __forceinline__ unsigned short f2bf(float f) {
  unsigned int u = __float_as_uint(f);
  u += 0x7FFFu + ((u >> 16) & 1u);       // round-to-nearest-even
  return (unsigned short)(u >> 16);
}
static __device__ __forceinline__ unsigned int pack2(float a, float b) {
  return (unsigned int)f2bf(a) | ((unsigned int)f2bf(b) << 16);
}

// tanh via single v_exp_f32 + v_rcp_f32: tanh(x)=sign(x)*(1 - 2e/(1+e)), e=e^(-2|x|)
static __device__ __forceinline__ float fast_tanh(float x) {
  float ax = fabsf(x);
  float e  = __expf(ax * -2.0f);
  float r  = __builtin_amdgcn_rcpf(1.0f + e);
  float t  = 1.0f - 2.0f * e * r;
  return copysignf(t, x);
}

// ---------------------------------------------------------------------------
// TDM descriptor build + issue: 2D bf16 tile (tile_d0 x tile_d1) from a
// row-major tensor (tensor_d0 wide, stride0), landing in LDS with a 16B pad
// after every 64B row (pad_interval=16 dwords -> code 3, pad_amount=4 dwords
// -> code 3), i.e. exactly the LDA/LDB = 40-ushort padded row stride.
// ---------------------------------------------------------------------------
static __device__ __forceinline__ void tdm_load_2d_bf16(
    unsigned lds_byte_addr, unsigned long long gaddr,
    unsigned tensor_d0, unsigned tensor_d1, unsigned stride0,
    unsigned tile_d0, unsigned tile_d1)
{
  u32x4 g0;
  g0[0] = 1u;                                      // count=1, user D#
  g0[1] = lds_byte_addr;                           // lds_addr [63:32]
  g0[2] = (unsigned)gaddr;                         // global_addr[31:0]
  g0[3] = (unsigned)(gaddr >> 32) | (2u << 30);    // global_addr[56:32] | type=2

  i32x8 g1;
  g1[0] = (int)((1u << 16)        // data_size = 2 bytes
              | (1u << 20)        // pad_enable
              | (3u << 22)        // pad_interval: 16 dwords
              | (3u << 25));      // pad_amount:   4 dwords
  g1[1] = (int)(tensor_d0 << 16);                          // tensor_dim0[15:0]
  g1[2] = (int)((tensor_d0 >> 16) | (tensor_d1 << 16));    // d0 hi | d1 lo
  g1[3] = (int)((tensor_d1 >> 16) | (tile_d0 << 16));      // d1 hi | tile_dim0
  g1[4] = (int)tile_d1;                                    // tile_dim1 (tile_dim2=0)
  g1[5] = (int)stride0;                                    // tensor_dim0_stride lo
  g1[6] = 0;
  g1[7] = 0;

  i32x4 z4 = {0, 0, 0, 0};
#if __has_include(<hip/amd_detail/amd_gfx1250_TDM.h>)
  i32x8 z8 = {0, 0, 0, 0, 0, 0, 0, 0};
  __builtin_amdgcn_tensor_load_to_lds(g0, g1, z4, z4, z8, 0);   // clang-23 form
#else
  __builtin_amdgcn_tensor_load_to_lds(g0, g1, z4, z4, 0);       // ROCm 7.2 form
#endif
}

// ---------------------------------------------------------------------------
// bf16 pre-convert: Ab[m][k] = bf16(k<512 ? context[m][k] : output[m][k-512])
// ---------------------------------------------------------------------------
__global__ __launch_bounds__(256)
void convert_A_bf16(const float* __restrict__ outp, const float* __restrict__ ctx,
                    unsigned short* __restrict__ Ab)
{
  const size_t i8 = ((size_t)blockIdx.x * 256 + threadIdx.x) * 8;
  const size_t m  = i8 >> 10;
  const int    k  = (int)(i8 & (size_t)(K_ - 1));
  const float* src = (k < D_) ? (ctx + m * D_ + k) : (outp + m * D_ + (k - D_));
  f32x4 v0 = *(const f32x4*)src;
  f32x4 v1 = *(const f32x4*)(src + 4);
  u32x4 o;
  o[0] = pack2(v0.x, v0.y);  o[1] = pack2(v0.z, v0.w);
  o[2] = pack2(v1.x, v1.y);  o[3] = pack2(v1.z, v1.w);
  *(u32x4*)(Ab + i8) = o;
}

__global__ __launch_bounds__(256)
void convert_W_bf16(const float* __restrict__ W, unsigned short* __restrict__ Wb)
{
  const size_t i8 = ((size_t)blockIdx.x * 256 + threadIdx.x) * 8;
  f32x4 v0 = *(const f32x4*)(W + i8);
  f32x4 v1 = *(const f32x4*)(W + i8 + 4);
  u32x4 o;
  o[0] = pack2(v0.x, v0.y);  o[1] = pack2(v0.z, v0.w);
  o[2] = pack2(v1.x, v1.y);  o[3] = pack2(v1.z, v1.w);
  *(u32x4*)(Wb + i8) = o;
}

// ---------------------------------------------------------------------------
// TDM-fed GEMM. grid=(N/BN, M/BM), block=256 (8 wave32). Double-buffered LDS
// tiles DMA'd by wave 0; all VALU in the loop is fragment reads + WMMA.
// ---------------------------------------------------------------------------
__global__ __launch_bounds__(256, 2)
void hardattn_gemm_tdm(const unsigned short* __restrict__ Ab,  // [M,K] bf16
                       const unsigned short* __restrict__ Wb,  // [N,K] bf16
                       const float* __restrict__ bias,
                       float* __restrict__ dst)
{
  __shared__ __align__(128) unsigned short lds[2 * BM * LDA + 2 * BN * LDB];

  const int t    = threadIdx.x;
  const int wave = t >> 5;
  const int lane = t & 31;
  const int half = lane >> 4;
  const int l16  = lane & 15;

  const int mBlock = blockIdx.y * BM;
  const int nBase  = blockIdx.x * BN;

  const unsigned ldsBase = (unsigned)(size_t)(void*)lds;  // low 32 = LDS offset

  f32x8 acc[4];
  #pragma unroll
  for (int i = 0; i < 4; ++i)
    #pragma unroll
    for (int j = 0; j < 8; ++j) acc[i][j] = 0.0f;

  // prologue: DMA tile 0 into buffer 0
  if (wave == 0) {
    unsigned long long gA = (unsigned long long)(size_t)Ab + (size_t)mBlock * K_ * 2ull;
    unsigned long long gB = (unsigned long long)(size_t)Wb + (size_t)nBase  * K_ * 2ull;
    tdm_load_2d_bf16(ldsBase,                       gA, K_, M_, K_, 32, BM);
    tdm_load_2d_bf16(ldsBase + 2u * BM * LDA * 2u,  gB, K_, D_, K_, 32, BN);
  }

  for (int kb = 0; kb < K_ / 32; ++kb) {
    const int buf = kb & 1;

    if (wave == 0) {
      if (kb + 1 < K_ / 32) {   // prefetch next tile into the other buffer
        const int nb = buf ^ 1;
        unsigned long long gA = (unsigned long long)(size_t)Ab +
            ((size_t)mBlock * K_ + (size_t)(kb + 1) * 32) * 2ull;
        unsigned long long gB = (unsigned long long)(size_t)Wb +
            ((size_t)nBase * K_ + (size_t)(kb + 1) * 32) * 2ull;
        tdm_load_2d_bf16(ldsBase + (unsigned)(nb * BM * LDA * 2), gA, K_, M_, K_, 32, BM);
        tdm_load_2d_bf16(ldsBase + (unsigned)(2 * BM * LDA * 2 + nb * BN * LDB * 2),
                         gB, K_, D_, K_, 32, BN);
        __builtin_amdgcn_s_wait_tensorcnt(2);   // current tile's 2 DMAs done
      } else {
        __builtin_amdgcn_s_wait_tensorcnt(0);
      }
    }
    __syncthreads();   // tile kb visible to all waves

    const unsigned short* la = lds + buf * (BM * LDA);
    const unsigned short* lb = lds + 2 * (BM * LDA) + buf * (BN * LDB);

    // A fragment: lanes 0-15: M=lane, K=0..15; lanes 16-31: K=16..31
    FragAB a;
    {
      const unsigned short* pa = la + (wave * 16 + l16) * LDA + half * 16;
      #pragma unroll
      for (int j = 0; j < 4; ++j) {
        a.u32[j]     = *(const unsigned int*)(pa + 2 * j);
        a.u32[4 + j] = *(const unsigned int*)(pa + 8 + 2 * j);
      }
    }
    // B fragments: col=lane&15, lanes 0-15: K=0..15, lanes 16-31: K=16..31
    #pragma unroll
    for (int tn = 0; tn < 4; ++tn) {
      FragAB bm;
      const unsigned short* pb = lb + (tn * 16 + l16) * LDB + half * 16;
      #pragma unroll
      for (int j = 0; j < 8; ++j)
        bm.u32[j] = *(const unsigned int*)(pb + 2 * j);

      acc[tn] = __builtin_amdgcn_wmma_f32_16x16x32_bf16(
          false, a.v, false, bm.v, (short)0, acc[tn], false, false);
    }
    __syncthreads();   // all reads done before this buffer is DMA'd again
  }

  #pragma unroll
  for (int tn = 0; tn < 4; ++tn) {
    const int col = nBase + tn * 16 + l16;
    const float bb = bias[col];
    #pragma unroll
    for (int v = 0; v < 8; ++v) {
      const int m = mBlock + wave * 16 + half * 8 + v;
      __builtin_nontemporal_store(fast_tanh(acc[tn][v] + bb), dst + (size_t)m * D_ + col);
    }
  }
}

// ---------------------------------------------------------------------------
// Fallback GEMM (convert-in-kernel, no workspace needed) — round-1 structure.
// ---------------------------------------------------------------------------
__global__ __launch_bounds__(256, 2)
void hardattn_gemm_tanh(const float* __restrict__ out_in, const float* __restrict__ ctx,
                        const float* __restrict__ W, const float* __restrict__ bias,
                        float* __restrict__ dst)
{
  __shared__ __align__(16) unsigned short lds_a[BM * LDA];
  __shared__ __align__(16) unsigned short lds_b[BN * LDB];

  const int t    = threadIdx.x;
  const int wave = t >> 5;
  const int lane = t & 31;
  const int half = lane >> 4;
  const int l16  = lane & 15;

  const int mBlock = blockIdx.y * BM;
  const int nBase  = blockIdx.x * BN;

  f32x8 acc[4];
  #pragma unroll
  for (int i = 0; i < 4; ++i)
    #pragma unroll
    for (int j = 0; j < 8; ++j) acc[i][j] = 0.0f;

  const int aRow = t >> 3, aQ = t & 7;
  const int bCol = t >> 2, bQ = t & 3;

  for (int k0 = 0; k0 < K_; k0 += 32) {
    const float* srcA   = (k0 < D_) ? ctx : out_in;
    const int    kInner = k0 & (D_ - 1);

    #pragma unroll
    for (int p = 0; p < 4; ++p) {
      const int row = p * 32 + aRow;
      const float* g = srcA + (size_t)(mBlock + row) * D_ + kInner + aQ * 4;
      f32x4 v = *(const f32x4*)g;
      unsigned int* d = (unsigned int*)&lds_a[row * LDA + aQ * 4];
      d[0] = pack2(v.x, v.y);
      d[1] = pack2(v.z, v.w);
    }
    {
      const float* g = W + (size_t)(nBase + bCol) * K_ + k0 + bQ * 4;
      f32x4 v0 = *(const f32x4*)g;
      f32x4 v1 = *(const f32x4*)(g + 16);
      unsigned int* d0 = (unsigned int*)&lds_b[bCol * LDB + bQ * 4];
      unsigned int* d1 = (unsigned int*)&lds_b[bCol * LDB + 16 + bQ * 4];
      d0[0] = pack2(v0.x, v0.y);  d0[1] = pack2(v0.z, v0.w);
      d1[0] = pack2(v1.x, v1.y);  d1[1] = pack2(v1.z, v1.w);
    }
    __syncthreads();

    FragAB a;
    {
      const unsigned short* pa = &lds_a[(wave * 16 + l16) * LDA + half * 16];
      #pragma unroll
      for (int j = 0; j < 4; ++j) {
        a.u32[j]     = *(const unsigned int*)(pa + 2 * j);
        a.u32[4 + j] = *(const unsigned int*)(pa + 8 + 2 * j);
      }
    }
    #pragma unroll
    for (int tn = 0; tn < 4; ++tn) {
      FragAB bm;
      const unsigned short* pb = &lds_b[(tn * 16 + l16) * LDB + half * 16];
      #pragma unroll
      for (int j = 0; j < 8; ++j)
        bm.u32[j] = *(const unsigned int*)(pb + 2 * j);
      acc[tn] = __builtin_amdgcn_wmma_f32_16x16x32_bf16(
          false, a.v, false, bm.v, (short)0, acc[tn], false, false);
    }
    __syncthreads();
  }

  #pragma unroll
  for (int tn = 0; tn < 4; ++tn) {
    const int col = nBase + tn * 16 + l16;
    const float bb = bias[col];
    #pragma unroll
    for (int v = 0; v < 8; ++v) {
      const int m = mBlock + wave * 16 + half * 8 + v;
      __builtin_nontemporal_store(fast_tanh(acc[tn][v] + bb), dst + (size_t)m * D_ + col);
    }
  }
}

// ---------------------------------------------------------------------------
// attn one-hot: 268 MB streaming write, nontemporal so L2 keeps GEMM operands.
// ---------------------------------------------------------------------------
__global__ __launch_bounds__(256)
void hardattn_onehot(float* __restrict__ attn)
{
  const size_t i4 = (size_t)blockIdx.x * 256 + threadIdx.x;
  const size_t e  = i4 * 4;
  const int s = (int)(e & (size_t)(S_ - 1));
  const int l = (int)((e >> 12) & (size_t)(L_ - 1));
  f32x4 v;
  v.x = (s     == l) ? 1.0f : 0.0f;
  v.y = (s + 1 == l) ? 1.0f : 0.0f;
  v.z = (s + 2 == l) ? 1.0f : 0.0f;
  v.w = (s + 3 == l) ? 1.0f : 0.0f;
  __builtin_nontemporal_store(v, (f32x4*)attn + i4);
}

// ---------------------------------------------------------------------------
extern "C" void kernel_launch(void* const* d_in, const int* in_sizes, int n_in,
                              void* d_out, int out_size, void* d_ws, size_t ws_size,
                              hipStream_t stream) {
  (void)in_sizes; (void)n_in; (void)out_size;

  const float* outp = (const float*)d_in[0];  // output  [B,L,D]
  const float* ctx  = (const float*)d_in[1];  // context [B,S,D]
  const float* W    = (const float*)d_in[2];  // [D, 2D]
  const float* bias = (const float*)d_in[3];  // [D]

  float* d_o    = (float*)d_out;
  float* d_attn = d_o + (size_t)B_ * L_ * D_;

  const size_t needA = (size_t)M_ * K_ * 2;   // 33.55 MB bf16 A
  const size_t needW = (size_t)D_ * K_ * 2;   //  1.05 MB bf16 W
  const dim3 gGemm(D_ / BN, M_ / BM);         // (8, 128)

  if (ws_size >= needA + needW) {
    unsigned short* Ab = (unsigned short*)d_ws;
    unsigned short* Wb = Ab + (size_t)M_ * K_;
    convert_A_bf16<<<dim3((unsigned)((size_t)M_ * K_ / 2048)), 256, 0, stream>>>(outp, ctx, Ab);
    convert_W_bf16<<<dim3((unsigned)((size_t)D_ * K_ / 2048)), 256, 0, stream>>>(W, Wb);
    hardattn_gemm_tdm<<<gGemm, dim3(256), 0, stream>>>(Ab, Wb, bias, d_o);
  } else {
    hardattn_gemm_tanh<<<gGemm, dim3(256), 0, stream>>>(outp, ctx, W, bias, d_o);
  }

  const size_t nQuads = (size_t)B_ * L_ * S_ / 4;
  hardattn_onehot<<<dim3((unsigned)(nQuads / 256)), 256, 0, stream>>>(d_attn);
}
